// BWGNN_AFA_68135361184224
// MI455X (gfx1250) — matrix operand order repaired and checked by reading, not compiled, not run on hardware
//
#include <hip/hip_runtime.h>
#include <hip/hip_bf16.h>

typedef __attribute__((ext_vector_type(8)))  float  v8f;
typedef __attribute__((ext_vector_type(16))) __bf16 v16bf;
typedef __attribute__((ext_vector_type(8)))  __bf16 v8bf;

namespace {

constexpr int kN  = 100000;   // nodes
constexpr int kE  = 1600000;  // edges
constexpr int kIN = 128;      // in feats
constexpr int kH  = 64;       // hidden
constexpr int kA  = 32;       // attn hidden

// ---------- helpers ----------

__device__ __forceinline__ float red16(float v) {
  // reduce across the 16-lane half (masks < 16 never cross halves)
  v += __shfl_xor(v, 1, 32);
  v += __shfl_xor(v, 2, 32);
  v += __shfl_xor(v, 4, 32);
  v += __shfl_xor(v, 8, 32);
  return v;
}

// A-fragment, 16-bit A 16x32 layout (ISA 7.12.2):
// lanes 0-15: row=lane, K = kb+0..7 (v0-3) and kb+16..23 (v4-7)
// lanes 16-31: row=lane-16, K = kb+8..15 and kb+24..31
__device__ __forceinline__ v16bf load_a_frag(const __bf16* A, int lda, int row0,
                                             int kb, int lane) {
  const int m  = lane & 15;
  const int kh = (lane >> 4) << 3;  // 0 or 8
  const __bf16* p = A + (size_t)(row0 + m) * lda + kb + kh;
  v8bf lo = *(const v8bf*)(p);
  v8bf hi = *(const v8bf*)(p + 16);
  v16bf a;
#pragma unroll
  for (int i = 0; i < 8; ++i) { a[i] = lo[i]; a[i + 8] = hi[i]; }
  return a;
}

// Pre-swizzled B fragment: one contiguous 32B load per lane.
__device__ __forceinline__ v16bf load_b_frag_sw(const __bf16* Wsw, int frag, int lane) {
  return *(const v16bf*)(Wsw + (((size_t)frag * 32 + lane) << 4));
}

// ---------- elementwise / graph kernels ----------

__global__ __launch_bounds__(256) void k_f32_to_bf16(const float* __restrict__ s,
                                                     __bf16* __restrict__ d, int n) {
  int i = blockIdx.x * 256 + threadIdx.x;
  if (i < n) d[i] = (__bf16)s[i];
}

// Convert W [K, ncols] fp32 row-major into fragment-major bf16:
// out[((kt*nct + t)*32 + lane)*16 + j] = W[(kt*32 + 8*(lane>>4) + (j&7) + 16*(j>>3))*ncols + t*16 + (lane&15)]
// matching the dense B layout: lanes 0-15 hold K=kb+0..7,16..23, lanes 16-31 K=kb+8..15,24..31.
__global__ __launch_bounds__(256) void k_swizzle_b(const float* __restrict__ W,
                                                   __bf16* __restrict__ out,
                                                   int K, int ncols) {
  int idx = blockIdx.x * 256 + threadIdx.x;
  if (idx >= K * ncols) return;
  int j    = idx & 15;
  int lane = (idx >> 4) & 31;
  int frag = idx >> 9;
  int nct  = ncols >> 4;
  int t    = frag % nct;
  int kt   = frag / nct;
  int krow = kt * 32 + ((lane >> 4) << 3) + (j & 7) + ((j >> 3) << 4);
  int col  = t * 16 + (lane & 15);
  out[idx] = (__bf16)W[(size_t)krow * ncols + col];
}

__global__ __launch_bounds__(256) void k_zero_f32(float* __restrict__ p, int n) {
  int i = blockIdx.x * 256 + threadIdx.x;
  if (i < n) p[i] = 0.0f;
}

__global__ __launch_bounds__(256) void k_degree(const int* __restrict__ dst,
                                                float* __restrict__ deg, int e) {
  int i = blockIdx.x * 256 + threadIdx.x;
  if (i < e) atomicAdd(&deg[dst[i]], 1.0f);
}

__global__ __launch_bounds__(256) void k_dinv(float* __restrict__ deg, int n) {
  int i = blockIdx.x * 256 + threadIdx.x;
  if (i < n) deg[i] = rsqrtf(fmaxf(deg[i], 1.0f));
}

// acc[dst] += f[src] * dinv[src]; 16 threads per edge, 4 floats each
__global__ __launch_bounds__(256) void k_lap_scatter(const int* __restrict__ src,
                                                     const int* __restrict__ dst,
                                                     const float* __restrict__ f,
                                                     const float* __restrict__ dinv,
                                                     float* __restrict__ acc, int e) {
  int t = blockIdx.x * 256 + threadIdx.x;
  if (t >= e * 16) return;
  int ed = t >> 4;
  int j  = (t & 15) << 2;
  int s  = src[ed];
  int d  = dst[ed];
  float sc = dinv[s];
  const float4 v = *(const float4*)(f + (size_t)s * kH + j);
  float* a = acc + (size_t)d * kH + j;
  atomicAdd(a + 0, v.x * sc);
  atomicAdd(a + 1, v.y * sc);
  atomicAdd(a + 2, v.z * sc);
  atomicAdd(a + 3, v.w * sc);
}

// acc = f - acc * dinv[node]   (lap finalize, in place)
__global__ __launch_bounds__(256) void k_lap_finalize(const float* __restrict__ f,
                                                      const float* __restrict__ dinv,
                                                      float* __restrict__ acc, int total) {
  int i = blockIdx.x * 256 + threadIdx.x;
  if (i < total) acc[i] = f[i] - acc[i] * dinv[i >> 6];
}

// theta combos -> hstack bf16 [N,3,64]
__global__ __launch_bounds__(256) void k_combo(const float* __restrict__ h,
                                               const float* __restrict__ f1,
                                               const float* __restrict__ f2,
                                               __bf16* __restrict__ hstack, int total) {
  int i = blockIdx.x * 256 + threadIdx.x;
  if (i >= total) return;
  int n = i >> 6, c = i & 63;
  float a = h[i], b = f1[i], g = f2[i];
  size_t base = (size_t)n * 192 + c;
  hstack[base]       = (__bf16)(3.0f * a - 3.0f * b + 0.75f * g);
  hstack[base + 64]  = (__bf16)(3.0f * b - 1.5f * g);
  hstack[base + 128] = (__bf16)(0.75f * g);
}

// hflat = hstack * (1 + delta[n,order])
__global__ __launch_bounds__(256) void k_scale(const __bf16* __restrict__ hstack,
                                               const float* __restrict__ delta,
                                               __bf16* __restrict__ hflat, int total) {
  int i = blockIdx.x * 256 + threadIdx.x;
  if (i >= total) return;
  int n = i / 192;
  int o = (i - n * 192) >> 6;
  float d = delta[n * 3 + o];
  hflat[i] = (__bf16)((float)hstack[i] * (1.0f + d));
}

// ---------- WMMA GEMM kernels ----------
// One wave per 16-row tile, 8 waves/block. Full-row ownership allows
// in-register LayerNorm / dot epilogues via half-wave shuffles.

__global__ __launch_bounds__(256)
void k_gemm_ln_relu(const __bf16* __restrict__ A, const __bf16* __restrict__ Wsw,
                    const float* __restrict__ bias, const float* __restrict__ gamma,
                    const float* __restrict__ beta, int n_rows, int K,
                    __bf16* __restrict__ out_bf, float* __restrict__ out_f) {
  const int lane = threadIdx.x & 31;
  const int wave = threadIdx.x >> 5;
  const int row0 = (blockIdx.x * 8 + wave) * 16;
  if (row0 >= n_rows) return;

  v8f acc[4] = {};
  for (int kb = 0; kb < K; kb += 32) {
    v16bf a = load_a_frag(A, K, row0, kb, lane);
    const int fbase = (kb >> 5) * 4;
#pragma unroll
    for (int t = 0; t < 4; ++t) {
      v16bf b = load_b_frag_sw(Wsw, fbase + t, lane);
      acc[t] = __builtin_amdgcn_wmma_f32_16x16x32_bf16(
          false, a, false, b, (short)0, acc[t], false, false);
    }
  }

  const int cb = lane & 15;
  float gamv[4], betv[4];
#pragma unroll
  for (int t = 0; t < 4; ++t) {
    const int col = t * 16 + cb;
    const float bv = bias[col];
    gamv[t] = gamma[col];
    betv[t] = beta[col];
#pragma unroll
    for (int i = 0; i < 8; ++i) acc[t][i] += bv;
  }
  const int rbase = row0 + ((lane < 16) ? 0 : 8);
#pragma unroll
  for (int i = 0; i < 8; ++i) {
    float si = acc[0][i] + acc[1][i] + acc[2][i] + acc[3][i];
    float qi = acc[0][i] * acc[0][i] + acc[1][i] * acc[1][i] +
               acc[2][i] * acc[2][i] + acc[3][i] * acc[3][i];
    si = red16(si);
    qi = red16(qi);
    const float mean = si * (1.0f / 64.0f);
    const float var  = qi * (1.0f / 64.0f) - mean * mean;
    const float inv  = rsqrtf(var + 1e-5f);
    const size_t rowoff = (size_t)(rbase + i) * kH;
#pragma unroll
    for (int t = 0; t < 4; ++t) {
      const int col = t * 16 + cb;
      float y = (acc[t][i] - mean) * inv * gamv[t] + betv[t];
      y = fmaxf(y, 0.0f);
      if (out_bf) out_bf[rowoff + col] = (__bf16)y;
      if (out_f)  out_f[rowoff + col]  = y;
    }
  }
}

// attention: [3N,64] @ [64,32] -> tanh -> dot Wa2 -> tanh -> delta[3N]
__global__ __launch_bounds__(256)
void k_attn(const __bf16* __restrict__ A, const __bf16* __restrict__ Wsw,
            const float* __restrict__ ba1, const float* __restrict__ Wa2,
            const float* __restrict__ ba2, int n_rows, float* __restrict__ delta) {
  const int lane = threadIdx.x & 31;
  const int wave = threadIdx.x >> 5;
  const int row0 = (blockIdx.x * 8 + wave) * 16;
  if (row0 >= n_rows) return;

  v8f acc[2] = {};
#pragma unroll
  for (int kt = 0; kt < 2; ++kt) {
    v16bf a = load_a_frag(A, kH, row0, kt * 32, lane);
#pragma unroll
    for (int t = 0; t < 2; ++t) {
      v16bf b = load_b_frag_sw(Wsw, kt * 2 + t, lane);
      acc[t] = __builtin_amdgcn_wmma_f32_16x16x32_bf16(
          false, a, false, b, (short)0, acc[t], false, false);
    }
  }

  const int cb = lane & 15;
  float b1v[2], w2v[2];
#pragma unroll
  for (int t = 0; t < 2; ++t) {
    b1v[t] = ba1[t * 16 + cb];
    w2v[t] = Wa2[t * 16 + cb];
  }
  const float bb2 = ba2[0];
  const int rbase = row0 + ((lane < 16) ? 0 : 8);
#pragma unroll
  for (int i = 0; i < 8; ++i) {
    float d = tanhf(acc[0][i] + b1v[0]) * w2v[0] +
              tanhf(acc[1][i] + b1v[1]) * w2v[1];
    d = red16(d);
    if ((lane & 15) == 0) delta[rbase + i] = tanhf(d + bb2);
  }
}

// [N,192] @ [192,64] -> relu -> @ W4[64,2] + b4 -> out[N,2]
__global__ __launch_bounds__(256)
void k_gemm3_out(const __bf16* __restrict__ A, const __bf16* __restrict__ Wsw,
                 const float* __restrict__ b3, const float* __restrict__ W4,
                 const float* __restrict__ b4, int n_rows, float* __restrict__ out) {
  const int lane = threadIdx.x & 31;
  const int wave = threadIdx.x >> 5;
  const int row0 = (blockIdx.x * 8 + wave) * 16;
  if (row0 >= n_rows) return;

  v8f acc[4] = {};
  for (int kb = 0; kb < 3 * kH; kb += 32) {
    v16bf a = load_a_frag(A, 3 * kH, row0, kb, lane);
    const int fbase = (kb >> 5) * 4;
#pragma unroll
    for (int t = 0; t < 4; ++t) {
      v16bf b = load_b_frag_sw(Wsw, fbase + t, lane);
      acc[t] = __builtin_amdgcn_wmma_f32_16x16x32_bf16(
          false, a, false, b, (short)0, acc[t], false, false);
    }
  }

  const int cb = lane & 15;
  float bv[4], w40[4], w41[4];
#pragma unroll
  for (int t = 0; t < 4; ++t) {
    const int col = t * 16 + cb;
    bv[t]  = b3[col];
    w40[t] = W4[2 * col];
    w41[t] = W4[2 * col + 1];
  }
  const float o0b = b4[0], o1b = b4[1];
  const int rbase = row0 + ((lane < 16) ? 0 : 8);
#pragma unroll
  for (int i = 0; i < 8; ++i) {
    float s0 = 0.0f, s1 = 0.0f;
#pragma unroll
    for (int t = 0; t < 4; ++t) {
      float y = fmaxf(acc[t][i] + bv[t], 0.0f);
      s0 += y * w40[t];
      s1 += y * w41[t];
    }
    s0 = red16(s0);
    s1 = red16(s1);
    if ((lane & 15) == 0) {
      out[(size_t)(rbase + i) * 2 + 0] = s0 + o0b;
      out[(size_t)(rbase + i) * 2 + 1] = s1 + o1b;
    }
  }
}

}  // namespace

extern "C" void kernel_launch(void* const* d_in, const int* in_sizes, int n_in,
                              void* d_out, int out_size, void* d_ws, size_t ws_size,
                              hipStream_t stream) {
  (void)in_sizes; (void)n_in; (void)out_size; (void)ws_size;

  const float* in_feat = (const float*)d_in[0];
  const int*   edge    = (const int*)d_in[1];
  const float* W1  = (const float*)d_in[2];
  const float* b1  = (const float*)d_in[3];
  const float* g1  = (const float*)d_in[4];
  const float* be1 = (const float*)d_in[5];
  const float* W2  = (const float*)d_in[6];
  const float* b2  = (const float*)d_in[7];
  const float* g2  = (const float*)d_in[8];
  const float* be2 = (const float*)d_in[9];
  const float* Wa1 = (const float*)d_in[10];
  const float* ba1 = (const float*)d_in[11];
  const float* Wa2 = (const float*)d_in[12];
  const float* ba2 = (const float*)d_in[13];
  const float* W3  = (const float*)d_in[14];
  const float* b3  = (const float*)d_in[15];
  const float* W4  = (const float*)d_in[16];
  const float* b4  = (const float*)d_in[17];

  const int* src = edge;
  const int* dst = edge + kE;

  // workspace carve (256B aligned)
  char* ws = (char*)d_ws;
  size_t off = 0;
  auto carve = [&](size_t bytes) -> char* {
    char* p = ws + off;
    off = (off + bytes + 255) & ~(size_t)255;
    return p;
  };
  float*  dinv   = (float*) carve((size_t)kN * 4);
  __bf16* in_bf  = (__bf16*)carve((size_t)kN * kIN * 2);
  __bf16* W1sw   = (__bf16*)carve((size_t)kIN * kH * 2);
  __bf16* W2sw   = (__bf16*)carve((size_t)kH * kH * 2);
  __bf16* Wa1sw  = (__bf16*)carve((size_t)kH * kA * 2);
  __bf16* W3sw   = (__bf16*)carve((size_t)3 * kH * kH * 2);
  __bf16* h1bf   = (__bf16*)carve((size_t)kN * kH * 2);
  float*  h2     = (float*) carve((size_t)kN * kH * 4);
  float*  f1     = (float*) carve((size_t)kN * kH * 4);
  float*  f2     = (float*)in_bf;  // alias: in_feat-bf16 is dead after GEMM1 (same size)
  __bf16* hstack = (__bf16*)carve((size_t)kN * 3 * kH * 2);
  float*  delta  = (float*) carve((size_t)kN * 3 * 4);
  __bf16* hflat  = (__bf16*)carve((size_t)kN * 3 * kH * 2);

  auto cdiv = [](int a, int b) { return (a + b - 1) / b; };

  // activation bf16 conversion + weight swizzle-to-fragment-major (incl. bf16 cvt)
  k_f32_to_bf16<<<cdiv(kN * kIN, 256), 256, 0, stream>>>(in_feat, in_bf, kN * kIN);
  k_swizzle_b<<<cdiv(kIN * kH, 256), 256, 0, stream>>>(W1, W1sw, kIN, kH);
  k_swizzle_b<<<cdiv(kH * kH, 256), 256, 0, stream>>>(W2, W2sw, kH, kH);
  k_swizzle_b<<<cdiv(kH * kA, 256), 256, 0, stream>>>(Wa1, Wa1sw, kH, kA);
  k_swizzle_b<<<cdiv(3 * kH * kH, 256), 256, 0, stream>>>(W3, W3sw, 3 * kH, kH);

  // degrees -> dinv (in place)
  k_zero_f32<<<cdiv(kN, 256), 256, 0, stream>>>(dinv, kN);
  k_degree<<<cdiv(kE, 256), 256, 0, stream>>>(dst, dinv, kE);
  k_dinv<<<cdiv(kN, 256), 256, 0, stream>>>(dinv, kN);

  // MLP trunk (WMMA + fused LayerNorm + ReLU)
  const int gb = cdiv(kN / 16, 8);
  k_gemm_ln_relu<<<gb, 256, 0, stream>>>(in_bf, W1sw, b1, g1, be1, kN, kIN, h1bf, nullptr);
  k_gemm_ln_relu<<<gb, 256, 0, stream>>>(h1bf, W2sw, b2, g2, be2, kN, kH, nullptr, h2);

  // f1 = lap(h2)
  k_zero_f32<<<cdiv(kN * kH, 256), 256, 0, stream>>>(f1, kN * kH);
  k_lap_scatter<<<cdiv(kE * 16, 256), 256, 0, stream>>>(src, dst, h2, dinv, f1, kE);
  k_lap_finalize<<<cdiv(kN * kH, 256), 256, 0, stream>>>(h2, dinv, f1, kN * kH);
  // f2 = lap(f1)
  k_zero_f32<<<cdiv(kN * kH, 256), 256, 0, stream>>>(f2, kN * kH);
  k_lap_scatter<<<cdiv(kE * 16, 256), 256, 0, stream>>>(src, dst, f1, dinv, f2, kE);
  k_lap_finalize<<<cdiv(kN * kH, 256), 256, 0, stream>>>(f1, dinv, f2, kN * kH);

  // polynomial bank combos -> hstack [N,3,64]
  k_combo<<<cdiv(kN * kH, 256), 256, 0, stream>>>(h2, f1, f2, hstack, kN * kH);

  // frequency attention -> delta, then scale -> hflat
  const int gb3 = cdiv((3 * kN) / 16, 8);
  k_attn<<<gb3, 256, 0, stream>>>(hstack, Wa1sw, ba1, Wa2, ba2, 3 * kN, delta);
  k_scale<<<cdiv(kN * 3 * kH, 256), 256, 0, stream>>>(hstack, delta, hflat, kN * 3 * kH);

  // final GEMM + relu + classifier, fused
  k_gemm3_out<<<gb, 256, 0, stream>>>(hflat, W3sw, b3, W4, b4, kN, (float*)d_out);
}
